// RACLinearPure_57827439673842
// MI455X (gfx1250) — matrix operand order repaired and checked by reading, not compiled, not run on hardware
//
#include <hip/hip_runtime.h>

typedef __attribute__((ext_vector_type(16))) _Float16 v16h;
typedef __attribute__((ext_vector_type(8)))  _Float16 v8h;
typedef __attribute__((ext_vector_type(8)))  float    v8f;
typedef __attribute__((ext_vector_type(4)))  float    f32x4;

#define K_DIM   256
#define N_DIM   256
#define TILE_M  16
#define XS      264        // padded x-tile row stride (halves): 132 dwords -> conflict-free b128
#define THREADS 512        // 16 wave32 = 16 N-tiles
#define NUM_WGS 512

__global__ __launch_bounds__(THREADS) void
linear_wmma_wreg(const float* __restrict__ x,
                 const float* __restrict__ w,
                 const float* __restrict__ bias,
                 float* __restrict__ y,
                 int mtiles_total, int mtiles_per_wg)
{
    __shared__ _Float16 sx[2][TILE_M * XS];   // double-buffered f16 x tile (2 x 8.25 KB)

    const int tid  = threadIdx.x;
    const int lane = tid & 31;
    const int wave = tid >> 5;      // 0..15 : this wave's N-tile
    const int l16  = lane & 15;
    const int hi   = lane >> 4;     // half-wave
    const int n0   = wave * 16;

    // ---- A-fragments: W rows n0..n0+15, all K, f32->f16, kept in registers ----
    // A 16x32 f16 layout: lane&15 = row, K = kc*32 + 8*hi + {0..7}, +16 for elems 8..15
    v16h a[8];
    {
        const float* wrow = w + (n0 + l16) * K_DIM;
        #pragma unroll
        for (int kc = 0; kc < 8; ++kc) {
            const int kb = kc * 32 + hi * 8;
            f32x4 w0 = *(const f32x4*)(wrow + kb);
            f32x4 w1 = *(const f32x4*)(wrow + kb + 4);
            f32x4 w2 = *(const f32x4*)(wrow + kb + 16);
            f32x4 w3 = *(const f32x4*)(wrow + kb + 20);
            #pragma unroll
            for (int j = 0; j < 4; ++j) {
                a[kc][j]      = (_Float16)w0[j];
                a[kc][j + 4]  = (_Float16)w1[j];
                a[kc][j + 8]  = (_Float16)w2[j];
                a[kc][j + 12] = (_Float16)w3[j];
            }
        }
    }

    // D-tile rows are W-rows here: acc element r <-> output column n0 + 8*hi + r
    float bsr[8];
    #pragma unroll
    for (int r = 0; r < 8; ++r) bsr[r] = bias[n0 + hi * 8 + r];

    // staging assignment: thread -> (row, 8 consecutive floats) of the x tile
    const int srow = tid >> 5;
    const int scol = (lane) * 8;    // 0..248

    const int wg_first = blockIdx.x * mtiles_per_wg;
    int ntiles = mtiles_per_wg;
    if (wg_first + ntiles > mtiles_total) ntiles = mtiles_total - wg_first;
    if (ntiles <= 0) return;        // uniform per block

    // ---- prologue: stage tile 0 ----
    {
        const float* xs = x + ((long)wg_first * TILE_M + srow) * K_DIM + scol;
        f32x4 v0 = *(const f32x4*)(xs);
        f32x4 v1 = *(const f32x4*)(xs + 4);
        v8h h;
        #pragma unroll
        for (int j = 0; j < 4; ++j) { h[j] = (_Float16)v0[j]; h[j + 4] = (_Float16)v1[j]; }
        *(v8h*)(&sx[0][srow * XS + scol]) = h;
    }
    __syncthreads();

    int c = 0;
    for (int t = 0; t < ntiles; ++t) {
        const long m0 = (long)(wg_first + t) * TILE_M;
        const bool has_next = (t + 1) < ntiles;

        // issue next tile's global loads now; they stay in flight across compute
        f32x4 nx0, nx1;
        if (has_next) {
            const float* xs = x + (m0 + TILE_M + srow) * K_DIM + scol;
            nx0 = *(const f32x4*)(xs);
            nx1 = *(const f32x4*)(xs + 4);
        }

        // ---- B-fragments for all 8 K-chunks: issue all 16 ds_load_b128 up front ----
        // B 32x16 f16 layout: lane&15 = column (x row), K = kc*32 + 8*hi + {0..7}, +16 for elems 8..15
        const _Float16* sb = &sx[c][l16 * XS + hi * 8];
        v8h bl[8], bh[8];
        #pragma unroll
        for (int kc = 0; kc < 8; ++kc) {
            bl[kc] = *(const v8h*)(sb + kc * 32);
            bh[kc] = *(const v8h*)(sb + kc * 32 + 16);
        }

        v8f acc;
        #pragma unroll
        for (int r = 0; r < 8; ++r) acc[r] = bsr[r];

        #pragma unroll
        for (int kc = 0; kc < 8; ++kc) {
            v16h b;
            #pragma unroll
            for (int j = 0; j < 8; ++j) { b[j] = bl[kc][j]; b[j + 8] = bh[kc][j]; }
            acc = __builtin_amdgcn_wmma_f32_16x16x32_f16(
                /*neg_a=*/false, a[kc], /*neg_b=*/false, b,
                /*c_mod=*/(short)0, acc, /*reuse_a=*/false, /*reuse_b=*/false);
        }

        // ---- store: lane holds y[m0 + l16][n0 + 8*hi .. +7] -> 2 contiguous b128 ----
        {
            float* yp = y + (m0 + l16) * N_DIM + n0 + hi * 8;
            f32x4 o0, o1;
            #pragma unroll
            for (int j = 0; j < 4; ++j) { o0[j] = acc[j]; o1[j] = acc[j + 4]; }
            *(f32x4*)(yp)     = o0;
            *(f32x4*)(yp + 4) = o1;
        }

        // ---- convert + commit next tile into the other buffer ----
        if (has_next) {
            v8h h;
            #pragma unroll
            for (int j = 0; j < 4; ++j) { h[j] = (_Float16)nx0[j]; h[j + 4] = (_Float16)nx1[j]; }
            *(v8h*)(&sx[c ^ 1][srow * XS + scol]) = h;
        }
        __syncthreads();
        c ^= 1;
    }
}

extern "C" void kernel_launch(void* const* d_in, const int* in_sizes, int n_in,
                              void* d_out, int out_size, void* d_ws, size_t ws_size,
                              hipStream_t stream) {
    const float* x = (const float*)d_in[0];   // (16*8192, 256) f32
    const float* w = (const float*)d_in[1];   // (256, 256) f32
    const float* b = (const float*)d_in[2];   // (256,) f32
    float* y = (float*)d_out;                 // (16*8192, 256) f32

    const long rows   = (long)in_sizes[0] / K_DIM;    // 131072
    const int  mtiles = (int)(rows / TILE_M);         // 8192
    const int  per_wg = (mtiles + NUM_WGS - 1) / NUM_WGS;

    linear_wmma_wreg<<<dim3(NUM_WGS), dim3(THREADS), 0, stream>>>(
        x, w, b, y, mtiles, per_wg);
}
